// MSHGTModel_24919400251712
// MI455X (gfx1250) — compile-verified, matrix operand order checked by reading, not run on hardware
//
#include <hip/hip_runtime.h>
#include <math.h>

typedef __attribute__((ext_vector_type(16))) _Float16 v16h;
typedef __attribute__((ext_vector_type(8)))  float    v8f;

#define HIDF 128
#define DHEAD 64

__device__ __forceinline__ v8f wmma16(v16h a, v16h b, v8f c) {
  return __builtin_amdgcn_wmma_f32_16x16x32_f16(false, a, false, b, (short)0, c, false, false);
}

// K-index pattern for 16-bit A/B operand layout (ISA 7.12.2):
// lanes 0-15: elems 0-7 -> K 0..7, elems 8-15 -> K 16..23 ; lanes 16-31: +8
__device__ __forceinline__ int kpat(int e, int laneHi) { return (e < 8 ? e : e + 8) + laneHi * 8; }

// Load one 32-wide K-slice of a row-major f32 row as an f16 WMMA operand.
// Two contiguous 8-float runs -> 4x global_load_b128 + 8x v_cvt_pk_f16_f32.
__device__ __forceinline__ v16h load_row_f16(const float* __restrict__ p, int laneHi) {
  const float4* q0 = (const float4*)(p + laneHi * 8);
  const float4* q1 = (const float4*)(p + 16 + laneHi * 8);
  float4 a0 = q0[0], a1 = q0[1], b0 = q1[0], b1 = q1[1];
  v16h r;
  r[0]  = (_Float16)a0.x; r[1]  = (_Float16)a0.y; r[2]  = (_Float16)a0.z; r[3]  = (_Float16)a0.w;
  r[4]  = (_Float16)a1.x; r[5]  = (_Float16)a1.y; r[6]  = (_Float16)a1.z; r[7]  = (_Float16)a1.w;
  r[8]  = (_Float16)b0.x; r[9]  = (_Float16)b0.y; r[10] = (_Float16)b0.z; r[11] = (_Float16)b0.w;
  r[12] = (_Float16)b1.x; r[13] = (_Float16)b1.y; r[14] = (_Float16)b1.z; r[15] = (_Float16)b1.w;
  return r;
}

__device__ __forceinline__ float leaky(float x) { return x > 0.f ? x : 0.2f * x; }

// ---------------------------------------------------------------------------
// Pre-pack a weight matrix W[K,Nc] (f32 row-major) into f16 WMMA operand
// order: Ws[((kb*Nc + n)*32 + j)] = W[kb*32 + kpat(j&15, j>>4)][n].
// A lane (laneHi) then fetches its 16 B-operand halves as one 32B v16h load.
// ---------------------------------------------------------------------------
__global__ void pack_weight_kernel(const float* __restrict__ W, _Float16* __restrict__ Ws,
                                   int K, int Nc) {
  int idx = blockIdx.x * blockDim.x + threadIdx.x;
  if (idx >= K * Nc) return;
  int j = idx & 31;
  int rem = idx >> 5;
  int n = rem % Nc;
  int kb = rem / Nc;
  int jj = j & 15;
  int kk = kb * 32 + (jj < 8 ? jj : jj + 8) + (j >> 4) * 8;
  Ws[idx] = (_Float16)W[(size_t)kk * Nc + n];
}

// ---------------------------------------------------------------------------
// Generic WMMA GEMM: C = act(A[M,K] @ B[K,Nc] + bias + addm).
// One wave per 16x64 tile; B pre-packed f16 (2x b128 per sub-operand).
// Nc is a multiple of 64 at all call sites; only M is ragged (clamped rows,
// never stored).
// ---------------------------------------------------------------------------
__global__ void wmma_gemm(const float* __restrict__ A, int lda,
                          const _Float16* __restrict__ Bs, int Nc,
                          const float* __restrict__ bias,
                          const float* __restrict__ addm, int ldadd,
                          float* __restrict__ C, int ldc,
                          int M, int K, int relu) {
  const int lane = threadIdx.x & 31;
  const int laneHi = lane >> 4, laneLo = lane & 15;
  const int tm = blockIdx.y * 16, tn = blockIdx.x * 64;
  const int arow = min(tm + laneLo, M - 1);
  const float* arp = A + (size_t)arow * lda;
  v8f c0 = {}, c1 = {}, c2 = {}, c3 = {};
  for (int k0 = 0, kb = 0; k0 < K; k0 += 32, ++kb) {
    v16h a = load_row_f16(arp + k0, laneHi);
    const _Float16* bbase = Bs + ((size_t)kb * Nc + tn + laneLo) * 32 + laneHi * 16;
    v16h b0 = *(const v16h*)(bbase);
    v16h b1 = *(const v16h*)(bbase + 16 * 32);
    v16h b2 = *(const v16h*)(bbase + 32 * 32);
    v16h b3 = *(const v16h*)(bbase + 48 * 32);
    c0 = wmma16(a, b0, c0);
    c1 = wmma16(a, b1, c1);
    c2 = wmma16(a, b2, c2);
    c3 = wmma16(a, b3, c3);
  }
#pragma unroll
  for (int r = 0; r < 8; ++r) {
    int row = tm + r + laneHi * 8;
    if (row < M) {
      float v0 = c0[r], v1 = c1[r], v2 = c2[r], v3 = c3[r];
      if (bias) {
        const float* bp = bias + tn + laneLo;
        v0 += bp[0]; v1 += bp[16]; v2 += bp[32]; v3 += bp[48];
      }
      if (addm) {
        const float* ap = addm + (size_t)row * ldadd + tn + laneLo;
        v0 += ap[0]; v1 += ap[16]; v2 += ap[32]; v3 += ap[48];
      }
      if (relu) { v0 = fmaxf(v0, 0.f); v1 = fmaxf(v1, 0.f); v2 = fmaxf(v2, 0.f); v3 = fmaxf(v3, 0.f); }
      float* cp = C + (size_t)row * ldc + tn + laneLo;
      cp[0] = v0; cp[16] = v1; cp[32] = v2; cp[48] = v3;
    }
  }
}

// ---------------------------------------------------------------------------
// PE reconstruction loss, fused 16x64-tile kernel (s_pe never materialized):
// s = (Q@K^T)/128 ; accum += sum of max(s,0) - s*A + log1p(exp(-|s|))
// ---------------------------------------------------------------------------
__global__ void pe_loss_kernel(const float* __restrict__ Qm, const float* __restrict__ Km,
                               const float* __restrict__ Aorig, float* __restrict__ accum, int Nn) {
  const int lane = threadIdx.x & 31, laneHi = lane >> 4, laneLo = lane & 15;
  const int tm = blockIdx.y * 16, tn = blockIdx.x * 64;
  const float* ap = Qm + (size_t)min(tm + laneLo, Nn - 1) * HIDF;
  const float* bp0 = Km + (size_t)min(tn + laneLo,      Nn - 1) * HIDF;
  const float* bp1 = Km + (size_t)min(tn + 16 + laneLo, Nn - 1) * HIDF;
  const float* bp2 = Km + (size_t)min(tn + 32 + laneLo, Nn - 1) * HIDF;
  const float* bp3 = Km + (size_t)min(tn + 48 + laneLo, Nn - 1) * HIDF;
  v8f cacc[4] = {};
  for (int k0 = 0; k0 < HIDF; k0 += 32) {
    v16h a = load_row_f16(ap + k0, laneHi);
    cacc[0] = wmma16(a, load_row_f16(bp0 + k0, laneHi), cacc[0]);
    cacc[1] = wmma16(a, load_row_f16(bp1 + k0, laneHi), cacc[1]);
    cacc[2] = wmma16(a, load_row_f16(bp2 + k0, laneHi), cacc[2]);
    cacc[3] = wmma16(a, load_row_f16(bp3 + k0, laneHi), cacc[3]);
  }
  float local = 0.f;
#pragma unroll
  for (int r = 0; r < 8; ++r) {
    int row = tm + r + laneHi * 8;
    bool rok = row < Nn;
    const float* arow = Aorig + (size_t)min(row, Nn - 1) * Nn;
#pragma unroll
    for (int sub = 0; sub < 4; ++sub) {
      int col = tn + sub * 16 + laneLo;
      float s = cacc[sub][r] * (1.0f / HIDF);
      float aA = arow[min(col, Nn - 1)];
      float term = fmaxf(s, 0.f) - s * aA + log1pf(__expf(-fabsf(s)));
      local += (rok && col < Nn) ? term : 0.f;
    }
  }
#pragma unroll
  for (int off = 1; off < 32; off <<= 1) local += __shfl_xor(local, off, 32);
  if (lane == 0) atomicAdd(accum, local);
}

// ---------------------------------------------------------------------------
// Dense decoder: pred = sigmoid((outW @ outF^T)/128), 16x64 tiles.
// ---------------------------------------------------------------------------
__global__ void pred_kernel(const float* __restrict__ outW, const float* __restrict__ outF,
                            float* __restrict__ pred, int Nn) {
  const int lane = threadIdx.x & 31, laneHi = lane >> 4, laneLo = lane & 15;
  const int tm = blockIdx.y * 16, tn = blockIdx.x * 64;
  const float* ap = outW + (size_t)min(tm + laneLo, Nn - 1) * HIDF;
  const float* bp0 = outF + (size_t)min(tn + laneLo,      Nn - 1) * HIDF;
  const float* bp1 = outF + (size_t)min(tn + 16 + laneLo, Nn - 1) * HIDF;
  const float* bp2 = outF + (size_t)min(tn + 32 + laneLo, Nn - 1) * HIDF;
  const float* bp3 = outF + (size_t)min(tn + 48 + laneLo, Nn - 1) * HIDF;
  v8f cacc[4] = {};
  for (int k0 = 0; k0 < HIDF; k0 += 32) {
    v16h a = load_row_f16(ap + k0, laneHi);
    cacc[0] = wmma16(a, load_row_f16(bp0 + k0, laneHi), cacc[0]);
    cacc[1] = wmma16(a, load_row_f16(bp1 + k0, laneHi), cacc[1]);
    cacc[2] = wmma16(a, load_row_f16(bp2 + k0, laneHi), cacc[2]);
    cacc[3] = wmma16(a, load_row_f16(bp3 + k0, laneHi), cacc[3]);
  }
#pragma unroll
  for (int r = 0; r < 8; ++r) {
    int row = tm + r + laneHi * 8;
    if (row < Nn) {
      float* prow = pred + (size_t)row * Nn;
#pragma unroll
      for (int sub = 0; sub < 4; ++sub) {
        int col = tn + sub * 16 + laneLo;
        if (col < Nn) {
          float s = cacc[sub][r] * (1.0f / HIDF);
          prow[col] = 1.f / (1.f + __expf(-s));
        }
      }
    }
  }
}

// ---------------------------------------------------------------------------
// Flash attention, one wave per (16-row query tile, head). dh = 64.
// Streams 32-key tiles: S via 4 WMMAs, online softmax via shfl butterflies,
// P routed C-layout -> A-layout through per-wave LDS, PV via 4 WMMAs.
// Output scaled by log1p(deg)/l (degree scaler + softmax normalizer).
// ---------------------------------------------------------------------------
__global__ void attn_kernel(const float* __restrict__ Qm, const float* __restrict__ Km,
                            const float* __restrict__ Vm, const float* __restrict__ deg,
                            float* __restrict__ Out, int Nn) {
  __shared__ float pbuf[16 * 32];
  const int lane = threadIdx.x & 31, laneHi = lane >> 4, laneLo = lane & 15;
  const int tm = blockIdx.x * 16;
  const int co = blockIdx.y * DHEAD;

  const float* qp = Qm + (size_t)min(tm + laneLo, Nn - 1) * HIDF + co;
  v16h aq0 = load_row_f16(qp, laneHi);
  v16h aq1 = load_row_f16(qp + 32, laneHi);

  v8f acc0 = {}, acc1 = {}, acc2 = {}, acc3 = {};
  float mrow[8], lrow[8];
#pragma unroll
  for (int r = 0; r < 8; ++r) { mrow[r] = -1e30f; lrow[r] = 0.f; }

  for (int kt = 0; kt < Nn; kt += 32) {
    const float* kp0 = Km + (size_t)min(kt + laneLo,      Nn - 1) * HIDF + co;
    const float* kp1 = Km + (size_t)min(kt + 16 + laneLo, Nn - 1) * HIDF + co;
    v8f s0 = {}, s1 = {};
    s0 = wmma16(aq0, load_row_f16(kp0, laneHi), s0);
    s0 = wmma16(aq1, load_row_f16(kp0 + 32, laneHi), s0);
    s1 = wmma16(aq0, load_row_f16(kp1, laneHi), s1);
    s1 = wmma16(aq1, load_row_f16(kp1 + 32, laneHi), s1);

    const bool ok0 = (kt + laneLo) < Nn, ok1 = (kt + 16 + laneLo) < Nn;
#pragma unroll
    for (int r = 0; r < 8; ++r) {
      float v0 = ok0 ? s0[r] * 0.125f : -1e30f;   // 1/sqrt(64)
      float v1 = ok1 ? s1[r] * 0.125f : -1e30f;
      float t = fmaxf(v0, v1);
#pragma unroll
      for (int off = 1; off < 16; off <<= 1) t = fmaxf(t, __shfl_xor(t, off, 32));
      float nm = fmaxf(mrow[r], t);
      float corr = __expf(mrow[r] - nm);
      mrow[r] = nm;
      float p0 = __expf(v0 - nm), p1 = __expf(v1 - nm);
      float ps = p0 + p1;
#pragma unroll
      for (int off = 1; off < 16; off <<= 1) ps += __shfl_xor(ps, off, 32);
      lrow[r] = lrow[r] * corr + ps;
      acc0[r] *= corr; acc1[r] *= corr; acc2[r] *= corr; acc3[r] *= corr;
      pbuf[(r + laneHi * 8) * 32 + laneLo] = p0;          // row-major 16x32 stage
      pbuf[(r + laneHi * 8) * 32 + 16 + laneLo] = p1;
    }
    // read back in A-operand layout (DS ops are in-order within a wave)
    v16h pa;
#pragma unroll
    for (int e = 0; e < 16; ++e) {
      pa[e] = (_Float16)pbuf[laneLo * 32 + kpat(e, laneHi)];
    }
    v16h bv0, bv1, bv2, bv3;
#pragma unroll
    for (int e = 0; e < 16; ++e) {
      int kr = min(kt + kpat(e, laneHi), Nn - 1);   // p==0 for invalid keys
      const float* vp = Vm + (size_t)kr * HIDF + co + laneLo;
      bv0[e] = (_Float16)vp[0];
      bv1[e] = (_Float16)vp[16];
      bv2[e] = (_Float16)vp[32];
      bv3[e] = (_Float16)vp[48];
    }
    acc0 = wmma16(pa, bv0, acc0);
    acc1 = wmma16(pa, bv1, acc1);
    acc2 = wmma16(pa, bv2, acc2);
    acc3 = wmma16(pa, bv3, acc3);
  }
#pragma unroll
  for (int r = 0; r < 8; ++r) {
    int row = tm + r + laneHi * 8;
    if (row < Nn) {
      float sc = log1pf(deg[row]) / lrow[r];
      float* op = Out + (size_t)row * HIDF + co + laneLo;
      op[0]  = acc0[r] * sc;
      op[16] = acc1[r] * sc;
      op[32] = acc2[r] * sc;
      op[48] = acc3[r] * sc;
    }
  }
}

// ---------------------------------------------------------------------------
// GAT helpers (scatter/softmax over edges; inherently atomic)
// ---------------------------------------------------------------------------
__global__ void gat_logits(const float* __restrict__ h, const float* __restrict__ asrc,
                           const float* __restrict__ adst, float* __restrict__ es,
                           float* __restrict__ ed, int Nn, int H, int F) {
  int idx = blockIdx.x * blockDim.x + threadIdx.x;
  if (idx >= Nn * H) return;
  int n = idx / H, hh = idx - n * H;
  const float* hp = h + (size_t)n * H * F + hh * F;
  const float* ap = asrc + hh * F;
  const float* dp = adst + hh * F;
  float s = 0.f, d = 0.f;
  for (int f = 0; f < F; ++f) { float hv = hp[f]; s += hv * ap[f]; d += hv * dp[f]; }
  es[idx] = s; ed[idx] = d;
}

__global__ void edge_max_kernel(const int* __restrict__ src, const int* __restrict__ dst,
                                const float* __restrict__ es, const float* __restrict__ ed,
                                float* __restrict__ mMax, int Ec, int H) {
  int i = blockIdx.x * blockDim.x + threadIdx.x;
  if (i >= Ec * H) return;
  int e = i / H, hh = i - e * H;
  float v = leaky(es[src[e] * H + hh] + ed[dst[e] * H + hh]);
  float* a = &mMax[dst[e] * H + hh];
  if (v >= 0.f) atomicMax((int*)a, __float_as_int(v));
  else          atomicMin((unsigned int*)a, __float_as_uint(v));
}

__global__ void edge_sum_kernel(const int* __restrict__ src, const int* __restrict__ dst,
                                const float* __restrict__ es, const float* __restrict__ ed,
                                const float* __restrict__ mMax, float* __restrict__ den,
                                int Ec, int H) {
  int i = blockIdx.x * blockDim.x + threadIdx.x;
  if (i >= Ec * H) return;
  int e = i / H, hh = i - e * H;
  float v = leaky(es[src[e] * H + hh] + ed[dst[e] * H + hh]);
  atomicAdd(&den[dst[e] * H + hh], __expf(v - mMax[dst[e] * H + hh]));
}

__global__ void edge_scatter_kernel(const int* __restrict__ src, const int* __restrict__ dst,
                                    const float* __restrict__ es, const float* __restrict__ ed,
                                    const float* __restrict__ mMax, const float* __restrict__ den,
                                    const float* __restrict__ h, float* __restrict__ acc,
                                    int Ec, int H, int F) {
  int e = blockIdx.x;
  if (e >= Ec) return;
  int tid = threadIdx.x;
  int hh = tid / F, f = tid - hh * F;
  int s = src[e], d = dst[e];
  float v = leaky(es[s * H + hh] + ed[d * H + hh]);
  float alpha = __expf(v - mMax[d * H + hh]) / (den[d * H + hh] + 1e-16f);
  atomicAdd(&acc[(size_t)d * H * F + hh * F + f], h[(size_t)s * H * F + hh * F + f] * alpha);
}

// ---------------------------------------------------------------------------
// Misc small kernels
// ---------------------------------------------------------------------------
__global__ void fill_kernel(float* p, float v, int n) {
  int i = blockIdx.x * blockDim.x + threadIdx.x;
  if (i < n) p[i] = v;
}

__global__ void init_bias_sum(float* acc, const float* __restrict__ b, int total, int C, int T_) {
  int i = blockIdx.x * blockDim.x + threadIdx.x;
  if (i >= total) return;
  int j = i % C;
  float v = 0.f;
  for (int t = 0; t < T_; ++t) v += b[t * C + j];
  acc[i] = v;
}

__global__ void relu_kernel(float* p, int n) {
  int i = blockIdx.x * blockDim.x + threadIdx.x;
  if (i < n) p[i] = fmaxf(p[i], 0.f);
}

__global__ void link_loss_kernel(const float* __restrict__ pred, const int* __restrict__ i0,
                                 const int* __restrict__ i1, float* __restrict__ accum,
                                 int Pn, int Nn, int isNeg) {
  int i = blockIdx.x * blockDim.x + threadIdx.x;
  if (i >= Pn) return;
  float v = pred[(size_t)i0[i] * Nn + i1[i]];
  float lv = isNeg ? __logf(1.f - v) : __logf(v);
  lv = fmaxf(lv, -100.f);
  atomicAdd(accum, lv);
}

__global__ void finalize_kernel(float* out0, const float* __restrict__ lossv, int Pn, int Nn) {
  float pe = lossv[0] / ((float)Nn * (float)Nn);
  out0[0] = -(lossv[1] / (float)Pn) - (lossv[2] / (float)Pn) + pe;
}

// ---------------------------------------------------------------------------
// Host launcher
// ---------------------------------------------------------------------------
extern "C" void kernel_launch(void* const* d_in, const int* in_sizes, int n_in,
                              void* d_out, int out_size, void* d_ws, size_t ws_size,
                              hipStream_t stream) {
  (void)n_in; (void)out_size; (void)ws_size;
  const float* x     = (const float*)d_in[0];
  const int*   ei    = (const int*)d_in[1];
  const float* Aorig = (const float*)d_in[2];
  const int*   posI  = (const int*)d_in[3];
  const int*   negI  = (const int*)d_in[4];
  const float* deg   = (const float*)d_in[5];
  int pi = 6;
  const float* W_in      = (const float*)d_in[pi++];
  const float* a_src_in  = (const float*)d_in[pi++];
  const float* a_dst_in  = (const float*)d_in[pi++];
  const float* b_in      = (const float*)d_in[pi++];
  const float* W_out     = (const float*)d_in[pi++];
  const float* a_src_out = (const float*)d_in[pi++];
  const float* a_dst_out = (const float*)d_in[pi++];
  const float* b_out     = (const float*)d_in[pi++];
  const float* pe_Q      = (const float*)d_in[pi++];
  const float* pe_K      = (const float*)d_in[pi++];
  const float* trans_W1  = (const float*)d_in[pi++];
  const float* trans_b1  = (const float*)d_in[pi++];
  const float* trans_W2  = (const float*)d_in[pi++];
  const float* trans_b2  = (const float*)d_in[pi++];
  const float* blk_Wq    = (const float*)d_in[pi++];
  const float* blk_bq    = (const float*)d_in[pi++];
  const float* blk_Wk    = (const float*)d_in[pi++];
  const float* blk_bk    = (const float*)d_in[pi++];
  const float* blk_Wv    = (const float*)d_in[pi++];
  const float* blk_bv    = (const float*)d_in[pi++];
  const float* blk_Wo    = (const float*)d_in[pi++];
  const float* blk_bo    = (const float*)d_in[pi++];
  const float* lin_cat_W = (const float*)d_in[pi++];
  const float* lin_cat_b = (const float*)d_in[pi++];
  const float* mlp_W1    = (const float*)d_in[pi++];
  const float* mlp_b1    = (const float*)d_in[pi++];
  const float* mlp_W2    = (const float*)d_in[pi++];
  const float* mlp_b2    = (const float*)d_in[pi++];
  const float* pred_W    = (const float*)d_in[pi++];
  const float* pred_b    = (const float*)d_in[pi++];

  const int Nn = in_sizes[0] / 128;     // 5000
  const int Ec = in_sizes[1] / 6;       // 160000 (T=3, 2 rows)
  const int Pn = in_sizes[3] / 2;       // 100000

  float* out = (float*)d_out;
  float* ws = (float*)d_ws;
  auto cdiv = [](int a, int b) { return (a + b - 1) / b; };

  // --- f16 packed-weight arena at the front of ws (32B-aligned base) ---
  _Float16* wf16 = (_Float16*)ws;
  size_t f16off = 0;
  auto packW = [&](const float* W, int K, int Nc) {
    _Float16* p = wf16 + f16off;
    f16off += (size_t)K * Nc;
    pack_weight_kernel<<<cdiv(K * Nc, 256), 256, 0, stream>>>(W, p, K, Nc);
    return (const _Float16*)p;
  };
  const _Float16* pW_in[3], *pW_out[3], *pWq[2], *pWk[2], *pWv[2], *pWo[2];
  for (int t = 0; t < 3; ++t) pW_in[t]  = packW(W_in  + (size_t)t * 128 * 256, 128, 256);
  for (int t = 0; t < 3; ++t) pW_out[t] = packW(W_out + (size_t)t * 256 * 128, 256, 128);
  const _Float16* pTW1 = packW(trans_W1, 128, 256);
  const _Float16* pTW2 = packW(trans_W2, 256, 128);
  for (int i = 0; i < 2; ++i) {
    pWq[i] = packW(blk_Wq + (size_t)i * 128 * 128, 128, 128);
    pWk[i] = packW(blk_Wk + (size_t)i * 128 * 128, 128, 128);
    pWv[i] = packW(blk_Wv + (size_t)i * 128 * 128, 128, 128);
    pWo[i] = packW(blk_Wo + (size_t)i * 128 * 128, 128, 128);
  }
  const _Float16* pLC  = packW(lin_cat_W, 384, 128);
  const _Float16* pM1  = packW(mlp_W1, 128, 256);
  const _Float16* pM2  = packW(mlp_W2, 256, 128);
  const _Float16* pPW  = packW(pred_W, 128, 128);

  // --- float workspace after the f16 arena (keep 32B alignment) ---
  size_t off = (f16off + 15) / 2;  // f16 -> float units, padded
  off = (off + 7) & ~(size_t)7;
  auto alloc = [&](size_t n) { float* p = ws + off; off += (n + 7) & ~(size_t)7; return p; };
  float* h_t   = alloc((size_t)Nn * 256);
  float* es    = alloc((size_t)Nn * 2);
  float* ed    = alloc((size_t)Nn * 2);
  float* mMax  = alloc((size_t)Nn * 2);
  float* den   = alloc((size_t)Nn * 2);
  float* acc1  = alloc((size_t)Nn * 256);   // h1 (GAT layer1 out, ReLU'd)
  float* acc2  = alloc((size_t)Nn * 128);   // xg (GAT layer2 out)
  float* t256  = alloc((size_t)Nn * 256);   // shared 256-wide temp
  float* Qb    = alloc((size_t)Nn * 128);
  float* Kb    = alloc((size_t)Nn * 128);
  float* Vb    = alloc((size_t)Nn * 128);
  float* attnO = alloc((size_t)Nn * 128);
  float* xcat  = alloc((size_t)Nn * 384);   // [x0 | cur1 | cur2]
  float* xgt   = alloc((size_t)Nn * 128);
  float* outf  = alloc((size_t)Nn * 128);
  float* outW  = alloc((size_t)Nn * 128);
  float* lossv = alloc(4);                  // [pe_sum, pos_sum, neg_sum]

#define GEMM(A, lda, Bs, Ncol, bias, addm, ldadd, C, ldc, M, K, relu) \
  wmma_gemm<<<dim3((unsigned)cdiv(Ncol, 64), (unsigned)cdiv(M, 16)), 32, 0, stream>>>( \
      A, lda, Bs, Ncol, bias, addm, ldadd, C, ldc, M, K, relu)

  fill_kernel<<<1, 32, 0, stream>>>(lossv, 0.f, 4);

  // ---------------- GAT layer 1 (H=2, F=128, out 256) ----------------
  init_bias_sum<<<cdiv(Nn * 256, 256), 256, 0, stream>>>(acc1, b_in, Nn * 256, 256, 3);
  for (int t = 0; t < 3; ++t) {
    GEMM(x, 128, pW_in[t], 256, nullptr, nullptr, 0, h_t, 256, Nn, 128, 0);
    gat_logits<<<cdiv(Nn * 2, 256), 256, 0, stream>>>(h_t, a_src_in + (size_t)t * 2 * 128,
                                                      a_dst_in + (size_t)t * 2 * 128, es, ed, Nn, 2, 128);
    fill_kernel<<<cdiv(Nn * 2, 256), 256, 0, stream>>>(mMax, -1e30f, Nn * 2);
    fill_kernel<<<cdiv(Nn * 2, 256), 256, 0, stream>>>(den, 0.f, Nn * 2);
    const int* srcp = ei + (size_t)t * 2 * Ec;
    const int* dstp = srcp + Ec;
    edge_max_kernel<<<cdiv(Ec * 2, 256), 256, 0, stream>>>(srcp, dstp, es, ed, mMax, Ec, 2);
    edge_sum_kernel<<<cdiv(Ec * 2, 256), 256, 0, stream>>>(srcp, dstp, es, ed, mMax, den, Ec, 2);
    edge_scatter_kernel<<<Ec, 256, 0, stream>>>(srcp, dstp, es, ed, mMax, den, h_t, acc1, Ec, 2, 128);
  }
  relu_kernel<<<cdiv(Nn * 256, 256), 256, 0, stream>>>(acc1, Nn * 256);

  // ---------------- GAT layer 2 (H=1, F=128, out 128) ----------------
  init_bias_sum<<<cdiv(Nn * 128, 256), 256, 0, stream>>>(acc2, b_out, Nn * 128, 128, 3);
  for (int t = 0; t < 3; ++t) {
    GEMM(acc1, 256, pW_out[t], 128, nullptr, nullptr, 0, h_t, 128, Nn, 256, 0);
    gat_logits<<<cdiv(Nn, 256), 256, 0, stream>>>(h_t, a_src_out + (size_t)t * 128,
                                                  a_dst_out + (size_t)t * 128, es, ed, Nn, 1, 128);
    fill_kernel<<<cdiv(Nn, 256), 256, 0, stream>>>(mMax, -1e30f, Nn);
    fill_kernel<<<cdiv(Nn, 256), 256, 0, stream>>>(den, 0.f, Nn);
    const int* srcp = ei + (size_t)t * 2 * Ec;
    const int* dstp = srcp + Ec;
    edge_max_kernel<<<cdiv(Ec, 256), 256, 0, stream>>>(srcp, dstp, es, ed, mMax, Ec, 1);
    edge_sum_kernel<<<cdiv(Ec, 256), 256, 0, stream>>>(srcp, dstp, es, ed, mMax, den, Ec, 1);
    edge_scatter_kernel<<<Ec, 128, 0, stream>>>(srcp, dstp, es, ed, mMax, den, h_t, acc2, Ec, 1, 128);
  }

  // ---------------- PE reconstruction loss (fused) ----------------
  pe_loss_kernel<<<dim3((unsigned)cdiv(Nn, 64), (unsigned)cdiv(Nn, 16)), 32, 0, stream>>>(
      pe_Q, pe_K, Aorig, lossv + 0, Nn);

  // ---------------- GraphTransformer ----------------
  GEMM(acc2, 128, pTW1, 256, trans_b1, nullptr, 0, t256, 256, Nn, 128, 1);
  GEMM(t256, 256, pTW2, 128, trans_b2, nullptr, 0, xcat, 384, Nn, 256, 0);  // x0 -> xcat[:,0:128]

  for (int i = 0; i < 2; ++i) {
    const float* cur = xcat + (size_t)i * 128;
    GEMM(cur, 384, pWq[i], 128, blk_bq + i * 128, pe_Q, 128, Qb, 128, Nn, 128, 0);
    GEMM(cur, 384, pWk[i], 128, blk_bk + i * 128, pe_K, 128, Kb, 128, Nn, 128, 0);
    GEMM(cur, 384, pWv[i], 128, blk_bv + i * 128, nullptr, 0, Vb, 128, Nn, 128, 0);
    attn_kernel<<<dim3((unsigned)cdiv(Nn, 16), 2), 32, 0, stream>>>(Qb, Kb, Vb, deg, attnO, Nn);
    GEMM(attnO, 128, pWo[i], 128, blk_bo + i * 128, cur, 384,
         xcat + (size_t)(i + 1) * 128, 384, Nn, 128, 1);
  }

  GEMM(xcat, 384, pLC, 128, lin_cat_b, nullptr, 0, xgt, 128, Nn, 384, 1);

  // ---------------- Prediction MLP + decoder ----------------
  GEMM(xgt, 128, pM1, 256, mlp_b1, nullptr, 0, t256, 256, Nn, 128, 1);
  GEMM(t256, 256, pM2, 128, mlp_b2, nullptr, 0, outf, 128, Nn, 256, 0);
  GEMM(outf, 128, pPW, 128, pred_b, nullptr, 0, outW, 128, Nn, 128, 0);

  float* pred = out + 1;
  pred_kernel<<<dim3((unsigned)cdiv(Nn, 64), (unsigned)cdiv(Nn, 16)), 32, 0, stream>>>(outW, outf, pred, Nn);
  link_loss_kernel<<<cdiv(Pn, 256), 256, 0, stream>>>(pred, posI, posI + Pn, lossv + 1, Pn, Nn, 0);
  link_loss_kernel<<<cdiv(Pn, 256), 256, 0, stream>>>(pred, negI, negI + Pn, lossv + 2, Pn, Nn, 1);
  finalize_kernel<<<1, 1, 0, stream>>>(out, lossv, Pn, Nn);
#undef GEMM
}
